// HPN_RNNAgent_68513318306047
// MI455X (gfx1250) — matrix operand order repaired and checked by reading, not compiled, not run on hardware
//
#include <hip/hip_runtime.h>
#include <hip/hip_bf16.h>
#include <math.h>

typedef _Float16 f16;
typedef _Float16 v16h __attribute__((ext_vector_type(16)));
typedef _Float16 v8h  __attribute__((ext_vector_type(8)));
typedef _Float16 v4h  __attribute__((ext_vector_type(4)));
typedef float    v8f  __attribute__((ext_vector_type(8)));

#define BS_   1024
#define NAG   8
#define BA_   (BS_*NAG)     // 8192
#define NHH   128           // NH*H

__device__ __forceinline__ v8f wmma16(v16h a, v16h b, v8f c) {
  // D = A(16x32 f16) * B(32x16 f16) + C(16x16 f32)
  return __builtin_amdgcn_wmma_f32_16x16x32_f16(false, a, false, b, (short)0, c, false, false);
}
__device__ __forceinline__ v16h cat8(v8h lo, v8h hi) {
  return __builtin_shufflevector(lo, hi, 0,1,2,3,4,5,6,7,8,9,10,11,12,13,14,15);
}
__device__ __forceinline__ v8f zero8() {
  v8f z;
#pragma unroll
  for (int i = 0; i < 8; ++i) z[i] = 0.f;
  return z;
}
// fast sigmoid / tanh: one v_exp_f32 + one v_rcp_f32 each
__device__ __forceinline__ float fast_sigmoid(float g) {
  return __builtin_amdgcn_rcpf(1.f + __expf(-g));
}
__device__ __forceinline__ float fast_tanh(float u) {
  float t = __expf(2.f * u);
  return (t - 1.f) * __builtin_amdgcn_rcpf(t + 1.f);
}

// ---------------------------------------------------------------------------
// Pack hypernet W2 (+bias extension rows) into B-fragment-native f16 layout.
// K ordering is d-major: kk = d*64 + h  (so the GEMM A-fill needs only shifts).
// frag index = ((t*8 + c)*32 + lane)*16 + j ; maps to B'[kk, n] with
// kk = t*32 + 16*(lane/16) + j, n = c*16 + (lane%16).
// B'[d*64+h, n] = W2[h, d*128 + n]; rows [64D, 64D+D) carry bias b2[d*128+n].
// ---------------------------------------------------------------------------
template<int D, int KT>
__global__ void pack_hyper(const float* __restrict__ w2, const float* __restrict__ b2,
                           f16* __restrict__ bp) {
  const int total = KT * 8 * 512;
  for (int idx = blockIdx.x * blockDim.x + threadIdx.x; idx < total;
       idx += gridDim.x * blockDim.x) {
    int j  = idx & 15;
    int l  = (idx >> 4) & 31;
    int c  = (idx >> 9) & 7;
    int t  = idx >> 12;
    int kk = t * 32 + ((l >> 4) << 4) + j;
    int n  = c * 16 + (l & 15);
    float v = 0.f;
    if (kk < 64 * D) {
      int d = kk >> 6, h = kk & 63;
      v = w2[h * (D * NHH) + d * NHH + n];
    } else if (kk < 64 * D + D) {
      v = b2[(kk - 64 * D) * NHH + n];
    }
    bp[idx] = (f16)v;
  }
}

// Pack a plain 64x192 weight (gru_wi / gru_wh) into fragment layout (2 Ktiles, 12 Ntiles)
__global__ void pack_plain(const float* __restrict__ w, f16* __restrict__ bp) {
  const int total = 2 * 12 * 512;
  for (int idx = blockIdx.x * blockDim.x + threadIdx.x; idx < total;
       idx += gridDim.x * blockDim.x) {
    int j  = idx & 15;
    int l  = (idx >> 4) & 31;
    int cz = idx >> 9;            // t*12 + c
    int c  = cz % 12, t = cz / 12;
    int kk = t * 32 + ((l >> 4) << 4) + j;
    int n  = c * 16 + (l & 15);
    bp[idx] = (f16)w[kk * 192 + n];
  }
}

// ---------------------------------------------------------------------------
// Fused hypernet GEMM: per 64 (virtual) rows build A = [ef (x) h1 | ef | 0pad]
// (d-major K) in LDS as f16, multiply by packed B' (KPADx128) with WMMA,
// reduce the 8 rows per agent straight out of the C-fragment layout, write
// e_sum[BA,128]. NPER==7 (allies) pads each agent to 8 rows with ef=0.
// ---------------------------------------------------------------------------
template<int D, int NPER, int KPAD, int KT, bool ACCUM>
__global__ __launch_bounds__(128) void hyper_gemm(
    const float* __restrict__ ef_g, const float* __restrict__ w1,
    const float* __restrict__ b1, const f16* __restrict__ bp,
    float* __restrict__ out) {
  __shared__ float s_ef[64][D];
  __shared__ float s_w1[D][64];
  __shared__ float s_b1[64];
  __shared__ float s_h1[64][64];
  __shared__ f16 s_a[64][KPAD] __attribute__((aligned(16)));

  const int tid = threadIdx.x;
  const int vbase = blockIdx.x * 64;

  for (int i = tid; i < D * 64; i += 128) s_w1[i / 64][i % 64] = w1[i];
  if (tid < 64) s_b1[tid] = b1[tid];
  for (int i = tid; i < 64 * D; i += 128) {
    int r = i / D, d = i - r * D;
    int v = vbase + r;
    int real;
    if (NPER == 8) real = v;
    else { int b = v >> 3, a = v & 7; real = (a < NPER) ? (b * NPER + a) : -1; }
    s_ef[r][d] = (real >= 0) ? ef_g[(size_t)real * D + d] : 0.f;
  }
  __syncthreads();

  { // h1 = relu(ef @ W1 + b1): 2 threads per row, 32 outputs each
    int r  = tid >> 1;
    int hb = (tid & 1) * 32;
    float efr[D];
#pragma unroll
    for (int d = 0; d < D; ++d) efr[d] = s_ef[r][d];
    for (int h = 0; h < 32; ++h) {
      float acc = s_b1[hb + h];
#pragma unroll
      for (int d = 0; d < D; ++d) acc += efr[d] * s_w1[d][hb + h];
      s_h1[r][hb + h] = fmaxf(acc, 0.f);
    }
  }
  __syncthreads();

  // A tile fill, 4 halfs at a time (kk = d*64 + h; d,h via shifts only)
  {
    const int KQ = KPAD / 4;
    for (int i4 = tid; i4 < 64 * KQ; i4 += 128) {
      int r  = i4 / KQ;
      int kk = (i4 - r * KQ) * 4;
      v4h v;
      if (kk < 64 * D) {
        int d = kk >> 6, h = kk & 63;
        float e = s_ef[r][d];
#pragma unroll
        for (int u = 0; u < 4; ++u) v[u] = (f16)(e * s_h1[r][h + u]);
      } else {
#pragma unroll
        for (int u = 0; u < 4; ++u) {
          int d = kk + u - 64 * D;
          v[u] = (f16)((d < D) ? s_ef[r][d] : 0.f);
        }
      }
      *(v4h*)&s_a[r][kk] = v;
    }
  }
  __syncthreads();

  const int wave = tid >> 5;
  const int lane = tid & 31;
  const int row  = wave * 16 + (lane & 15);
  const int kb   = (lane >> 4) << 4;

  v8f acc[8];
#pragma unroll
  for (int c = 0; c < 8; ++c) acc[c] = zero8();

  for (int t = 0; t < KT; ++t) {
    const f16* ap = &s_a[row][t * 32 + kb];
    v16h a = cat8(*(const v8h*)ap, *(const v8h*)(ap + 8));
    const f16* bpt = bp + (size_t)t * 8 * 512 + lane * 16;
#pragma unroll
    for (int c = 0; c < 8; ++c) {
      v16h b = *(const v16h*)(bpt + c * 512);
      acc[c] = wmma16(a, b, acc[c]);
    }
  }

  // C layout: lanes 0-15 hold M=i (rows 0..7 = first agent of this 16-row tile),
  // lanes 16-31 hold M=i+8 (second agent). Sum over i = sum over that agent.
  const int bidx = (vbase + wave * 16) / 8 + (lane >> 4);
  const int ncol = lane & 15;
#pragma unroll
  for (int c = 0; c < 8; ++c) {
    float s = 0.f;
#pragma unroll
    for (int i = 0; i < 8; ++i) s += acc[c][i];
    float* p = &out[(size_t)bidx * NHH + c * 16 + ncol];
    if (ACCUM) *p = *p + s;   // exclusive ownership; prior kernel already stored
    else *p = s;
  }
}

// ---------------------------------------------------------------------------
// emb + merge + GRU (WMMA for x@Wi and h@Wh), writes hh to d_out.
// ---------------------------------------------------------------------------
__global__ __launch_bounds__(128) void gru_kernel(
    const float* __restrict__ own, const float* __restrict__ hidden,
    const int* __restrict__ aidx, const int* __restrict__ laidx,
    const float* __restrict__ fc1_w, const float* __restrict__ fc1_b,
    const float* __restrict__ agent_emb, const float* __restrict__ action_emb,
    const float* __restrict__ miw, const float* __restrict__ e_sum,
    const float* __restrict__ gbi, const float* __restrict__ gbh,
    const f16* __restrict__ bwi, const f16* __restrict__ bwh,
    float* __restrict__ hh_out) {
  __shared__ f16 sx[64][64] __attribute__((aligned(16)));
  __shared__ f16 sh[64][64] __attribute__((aligned(16)));
  __shared__ float s_hid[64][64];
  __shared__ float s_gate[64][128];
  __shared__ float s_fc1[640];
  __shared__ float s_fc1b[64];
  __shared__ float s_soft[128];
  __shared__ float s_bi[192];
  __shared__ float s_bh[192];

  const int tid = threadIdx.x;
  const int r0  = blockIdx.x * 64;

  for (int i = tid; i < 640; i += 128) s_fc1[i] = fc1_w[i];
  if (tid < 64) {
    s_fc1b[tid] = fc1_b[tid];
    float e0 = __expf(miw[tid]), e1 = __expf(miw[64 + tid]);
    float inv = __builtin_amdgcn_rcpf(e0 + e1);
    s_soft[tid] = e0 * inv;
    s_soft[64 + tid] = e1 * inv;
  }
  for (int i = tid; i < 192; i += 128) { s_bi[i] = gbi[i]; s_bh[i] = gbh[i]; }
  for (int i = tid; i < 64 * 64; i += 128) {
    int r = i >> 6, h = i & 63;
    float hv = hidden[(size_t)(r0 + r) * 64 + h];
    s_hid[r][h] = hv;
    sh[r][h] = (f16)hv;
  }
  __syncthreads();

  { // emb + merged inputs, relu -> x
    int r  = tid >> 1;
    int hb = (tid & 1) * 32;
    int grow = r0 + r;
    int ai = aidx[grow] * 64;
    int li = laidx[grow] * 64;
    float ow[10];
#pragma unroll
    for (int d = 0; d < 10; ++d) ow[d] = own[(size_t)grow * 10 + d];
    for (int h = 0; h < 32; ++h) {
      int col = hb + h;
      float acc = s_fc1b[col] + agent_emb[ai + col] + action_emb[li + col];
#pragma unroll
      for (int d = 0; d < 10; ++d) acc += ow[d] * s_fc1[d * 64 + col];
      acc += s_soft[col]      * e_sum[(size_t)grow * 128 + col]
           + s_soft[64 + col] * e_sum[(size_t)grow * 128 + 64 + col];
      sx[r][col] = (f16)fmaxf(acc, 0.f);
    }
  }
  __syncthreads();

  const int wave = tid >> 5;
  const int lane = tid & 31;
  const int mrow = wave * 16 + (lane & 15);
  const int kb   = (lane >> 4) << 4;
  const int ncol = lane & 15;
  const int hi8  = (lane >> 4) * 8;

  // pass 1: r/z gates (output tiles 0..7 of 192)
  for (int c = 0; c < 8; ++c) {
    v8f ga = zero8(), gb = zero8();
#pragma unroll
    for (int t = 0; t < 2; ++t) {
      const f16* px = &sx[mrow][t * 32 + kb];
      const f16* ph = &sh[mrow][t * 32 + kb];
      v16h ax = cat8(*(const v8h*)px, *(const v8h*)(px + 8));
      v16h ah = cat8(*(const v8h*)ph, *(const v8h*)(ph + 8));
      v16h bi_ = *(const v16h*)(bwi + ((t * 12 + c) * 32 + lane) * 16);
      v16h bh_ = *(const v16h*)(bwh + ((t * 12 + c) * 32 + lane) * 16);
      ga = wmma16(ax, bi_, ga);
      gb = wmma16(ah, bh_, gb);
    }
    int cc = c * 16 + ncol;
    float bsum = s_bi[cc] + s_bh[cc];
#pragma unroll
    for (int i = 0; i < 8; ++i) {
      s_gate[wave * 16 + hi8 + i][cc] = fast_sigmoid(ga[i] + gb[i] + bsum);
    }
  }
  __syncthreads();

  // pass 2: n candidate + blend (output tiles 8..11)
  for (int c = 8; c < 12; ++c) {
    v8f ga = zero8(), gb = zero8();
#pragma unroll
    for (int t = 0; t < 2; ++t) {
      const f16* px = &sx[mrow][t * 32 + kb];
      const f16* ph = &sh[mrow][t * 32 + kb];
      v16h ax = cat8(*(const v8h*)px, *(const v8h*)(px + 8));
      v16h ah = cat8(*(const v8h*)ph, *(const v8h*)(ph + 8));
      v16h bi_ = *(const v16h*)(bwi + ((t * 12 + c) * 32 + lane) * 16);
      v16h bh_ = *(const v16h*)(bwh + ((t * 12 + c) * 32 + lane) * 16);
      ga = wmma16(ax, bi_, ga);
      gb = wmma16(ah, bh_, gb);
    }
    int cc = c * 16 + ncol;
    int hcol = cc - 128;
#pragma unroll
    for (int i = 0; i < 8; ++i) {
      int lrow = wave * 16 + hi8 + i;
      float rg  = s_gate[lrow][hcol];
      float zg  = s_gate[lrow][64 + hcol];
      float inn = ga[i] + s_bi[cc];
      float hn  = gb[i] + s_bh[cc];
      float nn  = fast_tanh(inn + rg * hn);
      hh_out[(size_t)(r0 + lrow) * 64 + hcol] =
          (1.f - zg) * nn + zg * s_hid[lrow][hcol];
    }
  }
}

// ---------------------------------------------------------------------------
// q_normal + attack head. Uses M[b,j,k] = sum_h hh[b,h]*W2a[j,h*2+k] so the
// per-enemy contraction is only 64x2 per row.
// ---------------------------------------------------------------------------
__global__ __launch_bounds__(256) void out_kernel(
    const float* __restrict__ hh, const float* __restrict__ ef_g,
    const float* __restrict__ onw, const float* __restrict__ onb,
    const float* __restrict__ wa1, const float* __restrict__ ba1,
    const float* __restrict__ wa2, const float* __restrict__ ba2,
    const float* __restrict__ wb1, const float* __restrict__ bb1,
    const float* __restrict__ wb2, const float* __restrict__ bb2,
    const float* __restrict__ mow, float* __restrict__ q) {
  __shared__ float s_hh[32][64];
  __shared__ float s_M[32][128];
  __shared__ float s_b2[32][2];
  __shared__ float s_w2a[64 * 128];
  __shared__ float s_w1a[9 * 64];
  __shared__ float s_w1b[9 * 64];
  __shared__ float s_b1a[64], s_b1b[64];
  __shared__ float s_onw[64 * 6], s_onb[6];
  __shared__ float s_w2b[128];
  __shared__ float s_ba2[128];
  __shared__ float s_bb2[2];

  const int tid = threadIdx.x;
  const int b0  = blockIdx.x * 32;

  for (int i = tid; i < 64 * 128; i += 256) s_w2a[i] = wa2[i];
  for (int i = tid; i < 9 * 64; i += 256) { s_w1a[i] = wa1[i]; s_w1b[i] = wb1[i]; }
  for (int i = tid; i < 64 * 6; i += 256) s_onw[i] = onw[i];
  if (tid < 128) { s_w2b[tid] = wb2[tid]; s_ba2[tid] = ba2[tid]; }
  if (tid < 64) { s_b1a[tid] = ba1[tid]; s_b1b[tid] = bb1[tid]; }
  if (tid < 6) s_onb[tid] = onb[tid];
  if (tid < 2) s_bb2[tid] = bb2[tid];
  for (int i = tid; i < 32 * 64; i += 256)
    s_hh[i >> 6][i & 63] = hh[(size_t)(b0 + (i >> 6)) * 64 + (i & 63)];
  __syncthreads();

  // M[r][j*2+k]
  for (int idx = tid; idx < 32 * 128; idx += 256) {
    int r = idx >> 7, jk = idx & 127;
    int j = jk >> 1, k = jk & 1;
    const float* w = &s_w2a[j * 128 + k];
    float acc = 0.f;
#pragma unroll 8
    for (int h = 0; h < 64; ++h) acc += s_hh[r][h] * w[h * 2];
    s_M[r][jk] = acc;
  }
  if (tid < 64) { // bias2[r][k] = sum_h hh*hwa_b2[h*2+k]
    int r = tid >> 1, k = tid & 1;
    float acc = 0.f;
    for (int h = 0; h < 64; ++h) acc += s_hh[r][h] * s_ba2[h * 2 + k];
    s_b2[r][k] = acc;
  }
  if (tid < 192) { // q_normal
    int r = tid / 6, j = tid - r * 6;
    float acc = s_onb[j];
    for (int h = 0; h < 64; ++h) acc += s_hh[r][h] * s_onw[h * 6 + j];
    q[(size_t)(b0 + r) * 14 + j] = acc;
  }
  __syncthreads();

  { // attack head: one thread per (agent-row, enemy)
    int r = tid >> 3, e = tid & 7;
    int b = b0 + r;
    float ef[9];
#pragma unroll
    for (int d = 0; d < 9; ++d) ef[d] = ef_g[((size_t)b * 8 + e) * 9 + d];
    float q0 = 0.f, q1 = 0.f, a0 = 0.f, a1 = 0.f;
    for (int j = 0; j < 64; ++j) {
      float h1a = s_b1a[j], h1b = s_b1b[j];
#pragma unroll
      for (int d = 0; d < 9; ++d) {
        h1a += ef[d] * s_w1a[d * 64 + j];
        h1b += ef[d] * s_w1b[d * 64 + j];
      }
      h1a = fmaxf(h1a, 0.f);
      h1b = fmaxf(h1b, 0.f);
      q0 += h1a * s_M[r][j * 2];
      q1 += h1a * s_M[r][j * 2 + 1];
      a0 += h1b * s_w2b[j * 2];
      a1 += h1b * s_w2b[j * 2 + 1];
    }
    float qa0 = q0 + s_b2[r][0] + a0 + s_bb2[0];
    float qa1 = q1 + s_b2[r][1] + a1 + s_bb2[1];
    float e0 = __expf(mow[0]), e1 = __expf(mow[1]);
    float inv = __builtin_amdgcn_rcpf(e0 + e1);
    q[(size_t)b * 14 + 6 + e] = (e0 * inv) * qa0 + (e1 * inv) * qa1;
  }
}

// ---------------------------------------------------------------------------
extern "C" void kernel_launch(void* const* d_in, const int* in_sizes, int n_in,
                              void* d_out, int out_size, void* d_ws, size_t ws_size,
                              hipStream_t stream) {
  (void)in_sizes; (void)n_in; (void)out_size; (void)ws_size;
  const float* own    = (const float*)d_in[0];
  const float* ef     = (const float*)d_in[1];
  const float* af     = (const float*)d_in[2];
  const float* hidden = (const float*)d_in[3];
  const int*   aidx   = (const int*)d_in[4];
  const int*   laidx  = (const int*)d_in[5];
  const float* fc1_w  = (const float*)d_in[7];
  const float* fc1_b  = (const float*)d_in[8];
  const float* agent_emb  = (const float*)d_in[9];
  const float* action_emb = (const float*)d_in[10];
  const float* hw_en_w1 = (const float*)d_in[11];
  const float* hw_en_b1 = (const float*)d_in[12];
  const float* hw_en_w2 = (const float*)d_in[13];
  const float* hw_en_b2 = (const float*)d_in[14];
  const float* hw_al_w1 = (const float*)d_in[15];
  const float* hw_al_b1 = (const float*)d_in[16];
  const float* hw_al_w2 = (const float*)d_in[17];
  const float* hw_al_b2 = (const float*)d_in[18];
  const float* miw = (const float*)d_in[19];
  const float* gwi = (const float*)d_in[20];
  const float* gwh = (const float*)d_in[21];
  const float* gbi = (const float*)d_in[22];
  const float* gbh = (const float*)d_in[23];
  const float* onw = (const float*)d_in[24];
  const float* onb = (const float*)d_in[25];
  const float* wa1 = (const float*)d_in[26];
  const float* ba1 = (const float*)d_in[27];
  const float* wa2 = (const float*)d_in[28];
  const float* ba2 = (const float*)d_in[29];
  const float* wb1 = (const float*)d_in[30];
  const float* bb1 = (const float*)d_in[31];
  const float* wb2 = (const float*)d_in[32];
  const float* bb2 = (const float*)d_in[33];
  const float* mow = (const float*)d_in[34];

  char* ws = (char*)d_ws;
  float* e_sum = (float*)ws;                               // 8192*128*4 = 4 MB
  f16* bp_en = (f16*)(ws + 4194304);                       // 19*8*512 halfs
  f16* bp_al = (f16*)(ws + 4194304 + 155648);              // 21*8*512 halfs
  f16* bp_wi = (f16*)(ws + 4194304 + 155648 + 172032);     // 2*12*512 halfs
  f16* bp_wh = (f16*)(ws + 4194304 + 155648 + 172032 + 24576);

  float* qout = (float*)d_out;
  float* hh   = qout + (size_t)BS_ * NAG * 14;             // hh section of d_out

  pack_hyper<9, 19><<<76, 256, 0, stream>>>(hw_en_w2, hw_en_b2, bp_en);
  pack_hyper<10, 21><<<84, 256, 0, stream>>>(hw_al_w2, hw_al_b2, bp_al);
  pack_plain<<<48, 256, 0, stream>>>(gwi, bp_wi);
  pack_plain<<<48, 256, 0, stream>>>(gwh, bp_wh);

  // enemy: K = 64*9 + 9 -> pad 608 (19 Ktiles); plain store
  hyper_gemm<9, 8, 608, 19, false><<<1024, 128, 0, stream>>>(
      ef, hw_en_w1, hw_en_b1, bp_en, e_sum);
  // ally: K = 64*10 + 10 -> pad 672 (21 Ktiles); accumulate (exclusive ownership)
  hyper_gemm<10, 7, 672, 21, true><<<1024, 128, 0, stream>>>(
      af, hw_al_w1, hw_al_b1, bp_al, e_sum);

  gru_kernel<<<128, 128, 0, stream>>>(own, hidden, aidx, laidx, fc1_w, fc1_b,
                                      agent_emb, action_emb, miw, e_sum, gbi, gbh,
                                      bp_wi, bp_wh, hh);

  out_kernel<<<256, 256, 0, stream>>>(hh, ef, onw, onb, wa1, ba1, wa2, ba2,
                                      wb1, bb1, wb2, bb2, mow, qout);
}
